// ConvNeXtStyleBlock_56530359550551
// MI455X (gfx1250) — matrix operand order repaired and checked by reading, not compiled, not run on hardware
//
#include <hip/hip_runtime.h>
#include <hip/hip_bf16.h>

// ---------- types ----------
typedef __attribute__((ext_vector_type(16))) __bf16        v16bf;
typedef __attribute__((ext_vector_type(8)))  float         v8f;
typedef __attribute__((ext_vector_type(4)))  unsigned int  u32x4;
typedef __attribute__((ext_vector_type(2)))  unsigned int  u32x2;
typedef __attribute__((ext_vector_type(2)))  int           i32x2;   // b64 async payload type

union ABFrag { v16bf v; u32x4 q[2]; };

// CDNA5 async global->LDS path (ASYNCcnt), with safe fallback
#if defined(__has_builtin)
#  if __has_builtin(__builtin_amdgcn_global_load_async_to_lds_b64)
#    define USE_ASYNC_LDS 1
#  endif
#  if __has_builtin(__builtin_amdgcn_s_wait_asynccnt)
#    define WAIT_ASYNC() __builtin_amdgcn_s_wait_asynccnt(0)
#  endif
#endif
#ifndef USE_ASYNC_LDS
#  define USE_ASYNC_LDS 0
#endif
#ifndef WAIT_ASYNC
#  define WAIT_ASYNC() asm volatile("s_wait_asynccnt 0" ::: "memory")
#endif

__device__ __forceinline__ unsigned short f2bf(float f) {
  unsigned int u = __float_as_uint(f);
  u += 0x7FFFu + ((u >> 16) & 1u);   // round-to-nearest-even
  return (unsigned short)(u >> 16);
}

#define CCH 768           // channels
#define NROWS 16384       // 16*32*32 tokens

// ---------- weight f32 -> bf16 ----------
__global__ void wcvt_kernel(const float* __restrict__ w, unsigned short* __restrict__ wb, int n) {
  int i = blockIdx.x * blockDim.x + threadIdx.x;
  if (i < n) wb[i] = f2bf(w[i]);
}

// ---------- NCHW f32 -> row-major [n][c] bf16 (per-batch 768x1024 transpose) ----------
__global__ void tin_kernel(const float* __restrict__ x, unsigned short* __restrict__ xt) {
  __shared__ float tile[32][33];
  const int b = blockIdx.z;
  const float*          src = x  + (size_t)b * CCH * 1024;   // [768][1024]
  unsigned short*       dst = xt + (size_t)b * 1024 * CCH;   // [1024][768]
  int s = blockIdx.x * 32 + threadIdx.x;
#pragma unroll
  for (int i = 0; i < 4; i++) {
    int c = blockIdx.y * 32 + threadIdx.y + i * 8;
    tile[threadIdx.y + i * 8][threadIdx.x] = src[(size_t)c * 1024 + s];
  }
  __syncthreads();
  int c2 = blockIdx.y * 32 + threadIdx.x;
#pragma unroll
  for (int i = 0; i < 4; i++) {
    int s2 = blockIdx.x * 32 + threadIdx.y + i * 8;
    dst[(size_t)s2 * CCH + c2] = f2bf(tile[threadIdx.x][threadIdx.y + i * 8]);
  }
}

// ---------- row-major [n][o] f32 -> NCHW f32 ----------
__global__ void tout_kernel(const float* __restrict__ h, float* __restrict__ out) {
  __shared__ float tile[32][33];
  const int b = blockIdx.z;
  const float* src = h   + (size_t)b * 1024 * CCH;   // [1024][768]
  float*       dst = out + (size_t)b * CCH * 1024;   // [768][1024]
  int o = blockIdx.x * 32 + threadIdx.x;
#pragma unroll
  for (int i = 0; i < 4; i++) {
    int s = blockIdx.y * 32 + threadIdx.y + i * 8;
    tile[threadIdx.y + i * 8][threadIdx.x] = src[(size_t)s * CCH + o];
  }
  __syncthreads();
  int s2 = blockIdx.y * 32 + threadIdx.x;
#pragma unroll
  for (int i = 0; i < 4; i++) {
    int o2 = blockIdx.x * 32 + threadIdx.y + i * 8;
    dst[(size_t)o2 * 1024 + s2] = tile[threadIdx.x][threadIdx.y + i * 8];
  }
}

// ---------- fused GEMM (+bias, optional LayerNorm+GELU) ----------
// Block: 512 threads = 16 waves. Each block: 64 rows x all 768 cols.
// Wave w owns N-tiles [3w, 3w+3), all 4 M-tiles. K loop: 24 x (K=32 bf16 wmma).
template <bool DO_LN>
__global__ __launch_bounds__(512) void gemm_kernel(
    const unsigned short* __restrict__ A,     // [NROWS][768] bf16
    const unsigned short* __restrict__ Wb,    // [768][768] bf16, row o contiguous in k
    const float* __restrict__ bias,
    const float* __restrict__ gamma,
    const float* __restrict__ beta,
    unsigned short* __restrict__ outB,        // bf16 out (DO_LN)
    float* __restrict__ outF)                 // f32 out (!DO_LN)
{
  // A staging: 64 rows x 32 k bf16, row padded to 80B (40 ushort) -> conflict-free b128 reads
  __shared__ unsigned short ldsA[2][64 * 40];
  __shared__ float lds_sum[64];
  __shared__ float lds_sq[64];

  const int tid  = threadIdx.x;
  const int lane = tid & 31;
  const int wave = tid >> 5;
  const int half = lane >> 4;     // 0: lanes 0-15, 1: lanes 16-31
  const int lm   = lane & 15;
  const int n0   = blockIdx.x * 64;
  const int o0   = wave * 48;

  const int arow = tid >> 3;          // 0..63
  const int acol = (tid & 7) * 4;     // ushort offset 0..28 (8B per thread)

  const unsigned short* gA = A + (size_t)(n0 + arow) * CCH + acol;
  unsigned short* l0 = &ldsA[0][arow * 40 + acol];
  unsigned short* l1 = &ldsA[1][arow * 40 + acol];

#if USE_ASYNC_LDS
  typedef __attribute__((address_space(3))) i32x2 LDSV;
  const unsigned long long gA_u = (unsigned long long)(uintptr_t)gA;
  LDSV* lv0 = (LDSV*)(unsigned int)(uintptr_t)l0;
  LDSV* lv1 = (LDSV*)(unsigned int)(uintptr_t)l1;
  // prologue: stage K-block 0 via async DMA to LDS
  __builtin_amdgcn_global_load_async_to_lds_b64((i32x2*)(uintptr_t)gA_u, lv0, 0, 0);
  WAIT_ASYNC();
#else
  { u32x2 v = *(const u32x2*)gA; *(u32x2*)l0 = v; }
#endif
  __syncthreads();

  v8f acc[4][3];
#pragma unroll
  for (int mt = 0; mt < 4; mt++)
#pragma unroll
    for (int nt = 0; nt < 3; nt++)
#pragma unroll
      for (int r = 0; r < 8; r++) acc[mt][nt][r] = 0.0f;

  for (int kk = 0; kk < 24; kk++) {
    const int buf = kk & 1;
    // prefetch next A K-block into the other LDS buffer (barrier below orders reuse)
    if (kk < 23) {
#if USE_ASYNC_LDS
      __builtin_amdgcn_global_load_async_to_lds_b64(
          (i32x2*)(uintptr_t)(gA_u + (unsigned long long)(kk + 1) * 64u),
          (buf ? lv0 : lv1), 0, 0);
#else
      u32x2 v = *(const u32x2*)(gA + (kk + 1) * 32);
      *(u32x2*)(buf ? (void*)l0 : (void*)l1) = v;
#endif
    }

    // A fragments: lane<16 -> K{0..7,16..23}, lane>=16 -> K{8..15,24..31}  (ISA 16-bit A 16x32)
    ABFrag af[4];
#pragma unroll
    for (int mt = 0; mt < 4; mt++) {
      const unsigned short* p = &ldsA[buf][(mt * 16 + lm) * 40 + half * 8];
      af[mt].q[0] = *(const u32x4*)(p);
      af[mt].q[1] = *(const u32x4*)(p + 16);
    }

    // B fragments straight from L2-resident bf16 W: lane holds W[o][kg..kg+7] and W[o][kg+16..kg+23]
    const int kg = kk * 32 + half * 8;
#pragma unroll
    for (int nt = 0; nt < 3; nt++) {
      ABFrag bf;
      const unsigned short* q = Wb + (size_t)(o0 + nt * 16 + lm) * CCH + kg;
      bf.q[0] = *(const u32x4*)(q);
      bf.q[1] = *(const u32x4*)(q + 16);
#pragma unroll
      for (int mt = 0; mt < 4; mt++)
        acc[mt][nt] = __builtin_amdgcn_wmma_f32_16x16x32_bf16(
            false, af[mt].v, false, bf.v, (short)0, acc[mt][nt], false, false);
    }
#if USE_ASYNC_LDS
    if (kk < 23) WAIT_ASYNC();   // our async store to the other buffer must land pre-barrier
#endif
    __syncthreads();
  }

  // ---- epilogue ----
  float bo[3], go[3], bt[3];
#pragma unroll
  for (int nt = 0; nt < 3; nt++) {
    const int o = o0 + nt * 16 + lm;
    bo[nt] = bias[o];
    if (DO_LN) { go[nt] = gamma[o]; bt[nt] = beta[o]; }
  }
#pragma unroll
  for (int mt = 0; mt < 4; mt++)
#pragma unroll
    for (int nt = 0; nt < 3; nt++)
#pragma unroll
      for (int r = 0; r < 8; r++) acc[mt][nt][r] += bo[nt];

  if (DO_LN) {
    if (tid < 64) { lds_sum[tid] = 0.0f; lds_sq[tid] = 0.0f; }
    __syncthreads();
    // per-row partial sums: reduce 16 lanes (same row, 48 cols per wave), then LDS atomics
#pragma unroll
    for (int mt = 0; mt < 4; mt++) {
#pragma unroll
      for (int r = 0; r < 8; r++) {
        float s = acc[mt][0][r] + acc[mt][1][r] + acc[mt][2][r];
        float q = acc[mt][0][r] * acc[mt][0][r] + acc[mt][1][r] * acc[mt][1][r] +
                  acc[mt][2][r] * acc[mt][2][r];
#pragma unroll
        for (int m = 1; m <= 8; m <<= 1) {
          s += __shfl_xor(s, m, 32);
          q += __shfl_xor(q, m, 32);
        }
        if (lm == 0) {
          const int row = mt * 16 + r + half * 8;
          atomicAdd(&lds_sum[row], s);
          atomicAdd(&lds_sq[row], q);
        }
      }
    }
    __syncthreads();
#pragma unroll
    for (int mt = 0; mt < 4; mt++) {
#pragma unroll
      for (int r = 0; r < 8; r++) {
        const int row = mt * 16 + r + half * 8;
        const float mu  = lds_sum[row] * (1.0f / 768.0f);
        const float var = lds_sq[row] * (1.0f / 768.0f) - mu * mu;
        const float rs  = rsqrtf(var + 1e-6f);
        const size_t base = (size_t)(n0 + row) * CCH;
#pragma unroll
        for (int nt = 0; nt < 3; nt++) {
          float y = (acc[mt][nt][r] - mu) * rs * go[nt] + bt[nt];
          y = 0.5f * y * (1.0f + erff(y * 0.70710678118654752f));   // exact GELU
          outB[base + o0 + nt * 16 + lm] = f2bf(y);
        }
      }
    }
  } else {
#pragma unroll
    for (int mt = 0; mt < 4; mt++)
#pragma unroll
      for (int r = 0; r < 8; r++) {
        const int row = mt * 16 + r + half * 8;
        const size_t base = (size_t)(n0 + row) * CCH;
#pragma unroll
        for (int nt = 0; nt < 3; nt++)
          outF[base + o0 + nt * 16 + lm] = acc[mt][nt][r];
      }
  }
}

// ---------- launch ----------
extern "C" void kernel_launch(void* const* d_in, const int* in_sizes, int n_in,
                              void* d_out, int out_size, void* d_ws, size_t ws_size,
                              hipStream_t stream) {
  const float* x   = (const float*)d_in[0];
  const float* W1  = (const float*)d_in[1];
  const float* b1  = (const float*)d_in[2];
  const float* g1  = (const float*)d_in[3];
  const float* be1 = (const float*)d_in[4];
  const float* W2  = (const float*)d_in[5];
  const float* b2  = (const float*)d_in[6];
  const float* g2  = (const float*)d_in[7];
  const float* be2 = (const float*)d_in[8];
  const float* W3  = (const float*)d_in[9];
  const float* b3  = (const float*)d_in[10];
  float* out = (float*)d_out;

  const size_t BUF = (size_t)NROWS * CCH * 2;   // 25165824 B (bf16 [16384][768])
  char* ws = (char*)d_ws;
  unsigned short* XT = (unsigned short*)(ws);
  unsigned short* H1 = (unsigned short*)(ws + BUF);
  unsigned short* H2 = (unsigned short*)(ws + 2 * BUF);
  unsigned short* WB = (unsigned short*)(ws + 3 * BUF);
  // layer-3 f32 output aliases [XT|H1] (both dead by then): 50.3 MB
  float* H3 = (float*)(ws);

  const int wn = CCH * CCH;
  wcvt_kernel<<<(wn + 255) / 256, 256, 0, stream>>>(W1, WB, wn);
  wcvt_kernel<<<(wn + 255) / 256, 256, 0, stream>>>(W2, WB + wn, wn);
  wcvt_kernel<<<(wn + 255) / 256, 256, 0, stream>>>(W3, WB + 2 * wn, wn);

  dim3 tb(32, 8);
  tin_kernel<<<dim3(32, 24, 16), tb, 0, stream>>>(x, XT);

  gemm_kernel<true ><<<NROWS / 64, 512, 0, stream>>>(XT, WB,          b1, g1, be1, H1, nullptr);
  gemm_kernel<true ><<<NROWS / 64, 512, 0, stream>>>(H1, WB + wn,     b2, g2, be2, H2, nullptr);
  gemm_kernel<false><<<NROWS / 64, 512, 0, stream>>>(H2, WB + 2 * wn, b3, nullptr, nullptr,
                                                     nullptr, H3);

  tout_kernel<<<dim3(24, 32, 16), tb, 0, stream>>>(H3, out);
}
